// Net_7473243095259
// MI455X (gfx1250) — matrix operand order repaired and checked by reading, not compiled, not run on hardware
//
#include <hip/hip_runtime.h>
#include <hip/hip_bf16.h>
#include <cstddef>

typedef __attribute__((ext_vector_type(16))) __bf16 v16bf;
typedef __attribute__((ext_vector_type(8)))  float  v8f;

#define N_NODES 89250
#define IN_CH   500
#define IN_PAD  512
#define HID     256
#define OUT_CH  7

// ---------------------------------------------------------------------------
// zero fill (grid-stride)
// ---------------------------------------------------------------------------
__global__ void zero_f32(float* __restrict__ p, long n) {
    long i = (long)blockIdx.x * blockDim.x + threadIdx.x;
    long stride = (long)gridDim.x * blockDim.x;
    for (; i < n; i += stride) p[i] = 0.0f;
}

// ---------------------------------------------------------------------------
// W [K x 256] f32 (row-major)  ->  WT [256 x ldp] bf16, zero-padded cols >= K
// ---------------------------------------------------------------------------
__global__ void transpose_to_bf16(const float* __restrict__ W, int K, int ldp,
                                  __bf16* __restrict__ WT) {
    int j = blockIdx.y;                                   // 0..255: output col of W
    int k = blockIdx.x * blockDim.x + threadIdx.x;        // 0..ldp-1
    if (k >= ldp) return;
    float v = (k < K) ? W[(long)k * HID + j] : 0.0f;
    WT[(long)j * ldp + k] = (__bf16)v;
}

// ---------------------------------------------------------------------------
// scatter-add: agg[dst[e], f] += w[e] * x[src[e], f]
// one wave per edge, lanes stride over features (coalesced row access)
// ---------------------------------------------------------------------------
__global__ void scatter_edges(const float* __restrict__ x, int ldx, int F,
                              const int* __restrict__ src, const int* __restrict__ dst,
                              const float* __restrict__ w,
                              float* __restrict__ agg, int ldagg, int E) {
    int e = blockIdx.x * (blockDim.x >> 5) + (threadIdx.x >> 5);
    if (e >= E) return;
    int lane = threadIdx.x & 31;
    int s = src[e];
    int d = dst[e];
    float we = w[e];
    const float* xr = x + (long)s * ldx;
    float* ar = agg + (long)d * ldagg;
    for (int f = lane; f < F; f += 32) {
        __hip_atomic_fetch_add(ar + f, we * xr[f],
                               __ATOMIC_RELAXED, __HIP_MEMORY_SCOPE_AGENT);
    }
}

// ---------------------------------------------------------------------------
// A-fragment loaders: 16x32 bf16 A tile, one lane's 16 elements.
// lane<16 : row M=lane, K = {k0..k0+15}
// lane>=16: same rows,  K = {k0+16..k0+31}.
// ---------------------------------------------------------------------------
__device__ __forceinline__ v16bf load_a_v(const float* __restrict__ row,
                                          int k0, int hi) {
    int ka = k0 + hi * 16;
    float4 f0 = *(const float4*)(row + ka);
    float4 f1 = *(const float4*)(row + ka + 4);
    float4 f2 = *(const float4*)(row + ka + 8);
    float4 f3 = *(const float4*)(row + ka + 12);
    v16bf a;
    a[0]  = (__bf16)f0.x; a[1]  = (__bf16)f0.y; a[2]  = (__bf16)f0.z; a[3]  = (__bf16)f0.w;
    a[4]  = (__bf16)f1.x; a[5]  = (__bf16)f1.y; a[6]  = (__bf16)f1.z; a[7]  = (__bf16)f1.w;
    a[8]  = (__bf16)f2.x; a[9]  = (__bf16)f2.y; a[10] = (__bf16)f2.z; a[11] = (__bf16)f2.w;
    a[12] = (__bf16)f3.x; a[13] = (__bf16)f3.y; a[14] = (__bf16)f3.z; a[15] = (__bf16)f3.w;
    return a;
}

__device__ __forceinline__ v16bf load_a_m(const float* __restrict__ row,
                                          int k0, int hi, int kvalid) {
    int ka = k0 + hi * 16;
    int kb = ka + 8;
    v16bf a;
#pragma unroll
    for (int i = 0; i < 8; ++i) {
        int k1 = ka + i, k2 = kb + i;
        a[i]     = (__bf16)((k1 < kvalid) ? row[k1] : 0.0f);
        a[8 + i] = (__bf16)((k2 < kvalid) ? row[k2] : 0.0f);
    }
    return a;
}

// ---------------------------------------------------------------------------
// fused GraphConv GEMM:  C = relu(A1 @ B1 + A2 @ B2 + bias)
// A1/A2  : f32 [M x K], converted to bf16 in-register (vector path in hot loop)
// B1T/B2T: bf16 [256 x ldbt] transposed, K-padded weights
// One wave computes a 16x64 output strip (4 col tiles, 4 accumulators,
// 8 v_wmma_f32_16x16x32_bf16 per K-step). Block = 4 waves = 16x256 stripe.
// Tail K-step (layer-1 root term, K=500) handled once after the loop.
// ---------------------------------------------------------------------------
__global__ void gemm_graphconv_relu(
    const float* __restrict__ A1, int lda1,
    const float* __restrict__ A2, int lda2,
    const __bf16* __restrict__ B1T, const __bf16* __restrict__ B2T, int ldbt,
    const float* __restrict__ bias, float* __restrict__ C,
    int M, int kfull, int ktail, int kvalid2)
{
    int lane = threadIdx.x & 31;
    int tg   = threadIdx.x >> 5;            // col group 0..3  (64 cols each)
    int m0   = blockIdx.x * 16;
    if (m0 >= M) return;

    int hi = lane >> 4;
    int lr = lane & 15;

    int arow = m0 + lr;
    if (arow >= M) arow = M - 1;            // clamp (stores are guarded)
    const float* a1row = A1 + (long)arow * lda1;
    const float* a2row = A2 + (long)arow * lda2;

    int col0 = tg * 64 + lr;
    const __bf16* b1p = B1T + (long)col0 * ldbt + hi * 16;
    const __bf16* b2p = B2T + (long)col0 * ldbt + hi * 16;
    const long cs = 16L * ldbt;             // stride between col tiles

    v8f acc[4] = {{}, {}, {}, {}};

    for (int s = 0; s < kfull; ++s) {
        int k0 = s * 32;
        v16bf a1 = load_a_v(a1row, k0, hi);
        v16bf a2 = load_a_v(a2row, k0, hi);
#pragma unroll
        for (int j = 0; j < 4; ++j) {
            v16bf b1 = *(const v16bf*)(b1p + j * cs + k0);
            v16bf b2 = *(const v16bf*)(b2p + j * cs + k0);
            acc[j] = __builtin_amdgcn_wmma_f32_16x16x32_bf16(
                         false, a1, false, b1, (short)0, acc[j], false, false);
            acc[j] = __builtin_amdgcn_wmma_f32_16x16x32_bf16(
                         false, a2, false, b2, (short)0, acc[j], false, false);
        }
    }

    if (ktail) {                            // layer-1 only: K tail at k0=480
        int k0 = kfull * 32;
        v16bf a1 = load_a_v(a1row, k0, hi);            // agg is zero-padded
        v16bf a2 = load_a_m(a2row, k0, hi, kvalid2);   // x0: mask K>=500
#pragma unroll
        for (int j = 0; j < 4; ++j) {
            v16bf b1 = *(const v16bf*)(b1p + j * cs + k0);
            v16bf b2 = *(const v16bf*)(b2p + j * cs + k0);
            acc[j] = __builtin_amdgcn_wmma_f32_16x16x32_bf16(
                         false, a1, false, b1, (short)0, acc[j], false, false);
            acc[j] = __builtin_amdgcn_wmma_f32_16x16x32_bf16(
                         false, a2, false, b2, (short)0, acc[j], false, false);
        }
    }

    int rowbase = m0 + hi * 8;
#pragma unroll
    for (int j = 0; j < 4; ++j) {
        int col = col0 + j * 16;
        float bv = bias[col];
#pragma unroll
        for (int r = 0; r < 8; ++r) {
            int row = rowbase + r;
            if (row < M) {
                float v = acc[j][r] + bv;
                C[(long)row * HID + col] = v > 0.0f ? v : 0.0f;
            }
        }
    }
}

// ---------------------------------------------------------------------------
// head: logits = [x1|x2|x3] @ Wlin + blin ; log_softmax
// ---------------------------------------------------------------------------
__global__ void head_logsoftmax(const float* __restrict__ x1,
                                const float* __restrict__ x2,
                                const float* __restrict__ x3,
                                const float* __restrict__ Wlin,
                                const float* __restrict__ blin,
                                float* __restrict__ out, int M) {
    int n = blockIdx.x * blockDim.x + threadIdx.x;
    if (n >= M) return;
    float acc[OUT_CH];
#pragma unroll
    for (int j = 0; j < OUT_CH; ++j) acc[j] = blin[j];

    const float* xs[3] = { x1 + (long)n * HID, x2 + (long)n * HID, x3 + (long)n * HID };
#pragma unroll
    for (int sgm = 0; sgm < 3; ++sgm) {
        const float* xr = xs[sgm];
        const float* wr = Wlin + (long)sgm * HID * OUT_CH;
        for (int k = 0; k < HID; ++k) {
            float xv = xr[k];
#pragma unroll
            for (int j = 0; j < OUT_CH; ++j) acc[j] += xv * wr[k * OUT_CH + j];
        }
    }
    float m = acc[0];
#pragma unroll
    for (int j = 1; j < OUT_CH; ++j) m = fmaxf(m, acc[j]);
    float ssum = 0.0f;
#pragma unroll
    for (int j = 0; j < OUT_CH; ++j) ssum += __expf(acc[j] - m);
    float lse = __logf(ssum);
#pragma unroll
    for (int j = 0; j < OUT_CH; ++j) out[(long)n * OUT_CH + j] = acc[j] - m - lse;
}

// ---------------------------------------------------------------------------
extern "C" void kernel_launch(void* const* d_in, const int* in_sizes, int n_in,
                              void* d_out, int out_size, void* d_ws, size_t ws_size,
                              hipStream_t stream) {
    const float* x0   = (const float*)d_in[0];
    const int*   eidx = (const int*)  d_in[1];
    const float* ew   = (const float*)d_in[2];
    const float* W1r  = (const float*)d_in[3];
    const float* b1   = (const float*)d_in[4];
    const float* W1o  = (const float*)d_in[5];
    const float* W2r  = (const float*)d_in[6];
    const float* b2   = (const float*)d_in[7];
    const float* W2o  = (const float*)d_in[8];
    const float* W3r  = (const float*)d_in[9];
    const float* b3   = (const float*)d_in[10];
    const float* W3o  = (const float*)d_in[11];
    const float* Wlin = (const float*)d_in[12];
    const float* blin = (const float*)d_in[13];
    float* out = (float*)d_out;

    const int E = in_sizes[1] / 2;
    const int* src = eidx;
    const int* dst = eidx + E;

    // workspace carve-up
    float* agg = (float*)d_ws;                       // [N, 512] f32 (layer1) / [N,256] reuse
    float* x1  = agg + (long)N_NODES * IN_PAD;       // [N, 256] f32
    float* x2  = x1  + (long)N_NODES * HID;
    float* x3  = x2  + (long)N_NODES * HID;
    __bf16* w1rT = (__bf16*)(x3 + (long)N_NODES * HID);   // [256, 512] bf16
    __bf16* w1oT = w1rT + (long)HID * IN_PAD;
    __bf16* w2rT = w1oT + (long)HID * IN_PAD;             // [256, 256] bf16
    __bf16* w2oT = w2rT + (long)HID * HID;
    __bf16* w3rT = w2oT + (long)HID * HID;
    __bf16* w3oT = w3rT + (long)HID * HID;

    // --- weight transpose + bf16 convert (small) ---
    {
        dim3 g1((IN_PAD + 255) / 256, HID);
        transpose_to_bf16<<<g1, 256, 0, stream>>>(W1r, IN_CH, IN_PAD, w1rT);
        transpose_to_bf16<<<g1, 256, 0, stream>>>(W1o, IN_CH, IN_PAD, w1oT);
        dim3 g2((HID + 255) / 256, HID);
        transpose_to_bf16<<<g2, 256, 0, stream>>>(W2r, HID, HID, w2rT);
        transpose_to_bf16<<<g2, 256, 0, stream>>>(W2o, HID, HID, w2oT);
        transpose_to_bf16<<<g2, 256, 0, stream>>>(W3r, HID, HID, w3rT);
        transpose_to_bf16<<<g2, 256, 0, stream>>>(W3o, HID, HID, w3oT);
    }

    const int ntm = (N_NODES + 15) / 16;          // 5579 row-tile blocks
    const int edgeBlocks = (E + 7) / 8;           // 8 waves / 256-thread block

    // ---------------- layer 1 ----------------
    zero_f32<<<4096, 256, 0, stream>>>(agg, (long)N_NODES * IN_PAD);
    scatter_edges<<<edgeBlocks, 256, 0, stream>>>(x0, IN_CH, IN_CH, src, dst, ew,
                                                  agg, IN_PAD, E);
    gemm_graphconv_relu<<<ntm, 128, 0, stream>>>(
        agg, IN_PAD, x0, IN_CH,
        w1rT, w1oT, IN_PAD, b1, x1, N_NODES,
        /*kfull=*/15, /*ktail=*/1, /*kvalid2=*/IN_CH);

    // ---------------- layer 2 ----------------
    zero_f32<<<4096, 256, 0, stream>>>(agg, (long)N_NODES * HID);
    scatter_edges<<<edgeBlocks, 256, 0, stream>>>(x1, HID, HID, src, dst, ew,
                                                  agg, HID, E);
    gemm_graphconv_relu<<<ntm, 128, 0, stream>>>(
        agg, HID, x1, HID,
        w2rT, w2oT, HID, b2, x2, N_NODES,
        /*kfull=*/8, /*ktail=*/0, /*kvalid2=*/HID);

    // ---------------- layer 3 ----------------
    zero_f32<<<4096, 256, 0, stream>>>(agg, (long)N_NODES * HID);
    scatter_edges<<<edgeBlocks, 256, 0, stream>>>(x2, HID, HID, src, dst, ew,
                                                  agg, HID, E);
    gemm_graphconv_relu<<<ntm, 128, 0, stream>>>(
        agg, HID, x2, HID,
        w3rT, w3oT, HID, b3, x3, N_NODES,
        /*kfull=*/8, /*ktail=*/0, /*kvalid2=*/HID);

    // ---------------- head ----------------
    head_logsoftmax<<<(N_NODES + 255) / 256, 256, 0, stream>>>(
        x1, x2, x3, Wlin, blin, out, N_NODES);
}
